// BertSelfAttention_979252544303
// MI455X (gfx1250) — matrix operand order repaired and checked by reading, not compiled, not run on hardware
//
#include <hip/hip_runtime.h>

typedef __attribute__((ext_vector_type(16))) __bf16 v16bf;
typedef __attribute__((ext_vector_type(8)))  __bf16 v8bf;
typedef __attribute__((ext_vector_type(8)))  float  v8f;
typedef __attribute__((ext_vector_type(4)))  unsigned int u32x4;
typedef __attribute__((ext_vector_type(8)))  int i32x8;
typedef __attribute__((ext_vector_type(4)))  int i32x4;

#define SEQ    2048
#define BATCH  2
#define HID    1024
#define NH     16      // heads per batch
#define HD     64      // head dim
#define NHEADS 32      // BATCH*NH
#define MROWS  4096    // SEQ*BATCH

#if defined(__has_builtin)
#  if __has_builtin(__builtin_amdgcn_tensor_load_to_lds) && \
      __has_builtin(__builtin_amdgcn_s_wait_tensorcnt)
#    define HAVE_TDM 1
#  endif
#endif
#ifndef HAVE_TDM
#  define HAVE_TDM 0
#endif

static __device__ __forceinline__ v16bf join8(v8bf lo, v8bf hi) {
  union { v16bf v; struct { v8bf lo, hi; } p; } u;
  u.p.lo = lo; u.p.hi = hi;
  return u.v;
}

static __device__ __forceinline__ v8f wmma_bf16(v16bf a, v16bf b, v8f c) {
  return __builtin_amdgcn_wmma_f32_16x16x32_bf16(false, a, false, b, (short)0, c,
                                                 false, false);
}

#if HAVE_TDM
// ---------------------------------------------------------------------------
// Tensor Data Mover: async 2D bf16 tile load Global -> LDS.
//   tile = tile_d1 rows x tile_d0 elements, row stride = stride0 elements.
//   D# layout per CDNA5 ISA (group0: count/lds/global/type,
//   group1: data_size=2B, tensor dims, tile dims, dim0 stride).
// ---------------------------------------------------------------------------
static __device__ __forceinline__ void tdm_load_2d(unsigned lds_off,
                                                   const void* gaddr,
                                                   unsigned tensor_d0,
                                                   unsigned tensor_d1,
                                                   unsigned tile_d0,
                                                   unsigned tile_d1,
                                                   unsigned stride0) {
  unsigned long long ga = (unsigned long long)gaddr;
  u32x4 g0;
  g0[0] = 1u;                                    // count=1 (valid), user mode
  g0[1] = lds_off;                               // lds_addr (bytes)
  g0[2] = (unsigned)(ga & 0xFFFFFFFFu);          // global_addr[31:0]
  g0[3] = (unsigned)((ga >> 32) & 0x01FFFFFFu) | 0x80000000u;  // [56:32]|type=2
  i32x8 g1;
  g1[0] = (int)(1u << 16);                       // data_size=1 -> 2 bytes
  g1[1] = (int)((tensor_d0 & 0xFFFFu) << 16);    // tensor_dim0[15:0] @ bit48
  g1[2] = (int)((tensor_d0 >> 16) | ((tensor_d1 & 0xFFFFu) << 16));
  g1[3] = (int)((tensor_d1 >> 16) | (tile_d0 << 16));   // tile_dim0 @ bit112
  g1[4] = (int)tile_d1;                          // tile_dim1 (tile_dim2 = 0)
  g1[5] = (int)stride0;                          // tensor_dim0_stride[31:0]
  g1[6] = 0;
  g1[7] = 0;
  i32x4 z4 = {0, 0, 0, 0};
#if __clang_major__ >= 23
  i32x8 z8 = {0, 0, 0, 0, 0, 0, 0, 0};
  __builtin_amdgcn_tensor_load_to_lds(g0, g1, z4, z4, z8, 0);
#else
  __builtin_amdgcn_tensor_load_to_lds(g0, g1, z4, z4, 0);
#endif
}

static __device__ __forceinline__ unsigned lds_off_of(const void* p) {
  // Generic pointers to LDS carry the LDS byte offset in addr[31:0]
  // (ISA 10.2).
  return (unsigned)(unsigned long long)p;
}
#endif  // HAVE_TDM

// ---------------------------------------------------------------------------
// f32 -> bf16 elementwise convert (hidden_states [S,B,H] flattens to X[4096,1024])
// ---------------------------------------------------------------------------
__global__ void cvt_bf16_kernel(const float* __restrict__ in,
                                __bf16* __restrict__ out, int n) {
  int i = blockIdx.x * blockDim.x + threadIdx.x;
  if (i < n) out[i] = (__bf16)in[i];
}

// ---------------------------------------------------------------------------
// W [K=1024][N=1024] f32 -> Wt [N][K] bf16 (N-major so the WMMA B-fragment
// load of 16 contiguous K-values is a single 32B global load)
// ---------------------------------------------------------------------------
__global__ void transpose_cvt_kernel(const float* __restrict__ W,
                                     __bf16* __restrict__ Wt) {
  __shared__ float tile[32][33];
  int kbase = blockIdx.x * 32;
  int nbase = blockIdx.y * 32;
#pragma unroll
  for (int i = 0; i < 4; ++i)
    tile[threadIdx.y + i * 8][threadIdx.x] =
        W[(size_t)(kbase + threadIdx.y + i * 8) * HID + nbase + threadIdx.x];
  __syncthreads();
#pragma unroll
  for (int i = 0; i < 4; ++i)
    Wt[(size_t)(nbase + threadIdx.y + i * 8) * HID + kbase + threadIdx.x] =
        (__bf16)tile[threadIdx.x][threadIdx.y + i * 8];
}

// ---------------------------------------------------------------------------
// QKV projection GEMM: X[4096,1024]bf16 x Wt[z][N][K]bf16 -> per-head bf16.
//   z=0 -> Qb[head][s][d]   z=1 -> Kb[head][s][d]   z=2 -> Vt[head][d][s]
// grid = (MROWS/16, HID/256, 3), block = 128 (4 waves, each wave a 16x64 tile)
// ---------------------------------------------------------------------------
__global__ __launch_bounds__(128) void qkv_gemm_kernel(
    const __bf16* __restrict__ X, const __bf16* __restrict__ Wt3,
    const float* __restrict__ bq, const float* __restrict__ bk,
    const float* __restrict__ bv, __bf16* __restrict__ Qb,
    __bf16* __restrict__ Kb, __bf16* __restrict__ Vt) {
  const int wave = threadIdx.x >> 5;
  const int lane = threadIdx.x & 31;
  const int hl   = lane >> 4;   // which 16-lane half
  const int l16  = lane & 15;
  const int z    = blockIdx.z;
  const int m0   = blockIdx.x * 16;
  const int n0   = blockIdx.y * 256 + wave * 64;

  const __bf16* W    = Wt3 + (size_t)z * HID * HID;
  const float*  bias = (z == 0) ? bq : ((z == 1) ? bk : bv);

  v8f acc[4] = {v8f{}, v8f{}, v8f{}, v8f{}};
  const int arow = m0 + l16;
  const __bf16* Arow = X + (size_t)arow * HID;

  for (int k = 0; k < HID; k += 32) {
    v8bf alo = *(const v8bf*)(Arow + k + hl * 16);
    v8bf ahi = *(const v8bf*)(Arow + k + hl * 16 + 8);
    v16bf a = join8(alo, ahi);
#pragma unroll
    for (int s = 0; s < 4; ++s) {
      int n = n0 + s * 16 + l16;
      v16bf b = *(const v16bf*)(W + (size_t)n * HID + k + hl * 16);
      acc[s] = wmma_bf16(a, b, acc[s]);
    }
  }

#pragma unroll
  for (int s = 0; s < 4; ++s) {
#pragma unroll
    for (int r = 0; r < 8; ++r) {
      int row  = m0 + r + hl * 8;      // row = s_idx*BATCH + b_idx
      int col  = n0 + s * 16 + l16;    // 0..1023
      float v  = acc[s][r] + bias[col];
      int sidx = row >> 1;             // BATCH == 2
      int bidx = row & 1;
      int h    = col >> 6;
      int d    = col & 63;
      int head = bidx * NH + h;
      __bf16 bvv = (__bf16)v;
      if (z == 2)
        Vt[((size_t)head * HD + d) * SEQ + sidx] = bvv;           // [head][d][s]
      else if (z == 0)
        Qb[((size_t)head * SEQ + sidx) * HD + d] = bvv;           // [head][s][d]
      else
        Kb[((size_t)head * SEQ + sidx) * HD + d] = bvv;
    }
  }
}

// ---------------------------------------------------------------------------
// Fused flash-attention. Block = 64 threads (2 waves), each wave owns 16 query
// rows; both waves share TDM-staged (double-buffered) K/V tiles in LDS.
// grid = (SEQ/32, NHEADS).
// ---------------------------------------------------------------------------
__global__ __launch_bounds__(64) void attn_kernel(
    const __bf16* __restrict__ Qb, const __bf16* __restrict__ Kb,
    const __bf16* __restrict__ Vt, const float* __restrict__ mask,
    float* __restrict__ out) {
  const int wave = threadIdx.x >> 5;
  const int lane = threadIdx.x & 31;
  const int hl   = lane >> 4;
  const int l16  = lane & 15;
  const int head = blockIdx.y;
  const int bidx = head >> 4;   // batch
  const int hidx = head & 15;   // head within batch
  const int m0   = blockIdx.x * 32 + wave * 16;

  const __bf16* Q = Qb + (size_t)head * SEQ * HD;
  const __bf16* K = Kb + (size_t)head * SEQ * HD;
  const __bf16* V = Vt + (size_t)head * HD * SEQ;
  const float* M  = mask + (size_t)bidx * SEQ * SEQ;   // [s][t]

  __shared__ __bf16 Kbuf[2][32 * HD];   // [t_local][d]   4KB each
  __shared__ __bf16 Vbuf[2][HD * 32];   // [d][t_local]   4KB each
  __shared__ __bf16 pls[2][16 * 32];    // per-wave P tile (C->A relayout)

  // Q fragments for d = 0..63 (two K=32 chunks), reused across all t.
  v16bf aq[2];
  {
    const __bf16* Qrow = Q + (size_t)(m0 + l16) * HD;
#pragma unroll
    for (int c = 0; c < 2; ++c) {
      v8bf lo = *(const v8bf*)(Qrow + c * 32 + hl * 16);
      v8bf hi = *(const v8bf*)(Qrow + c * 32 + hl * 16 + 8);
      aq[c] = join8(lo, hi);
    }
  }

  float mst[8], lst[8];
#pragma unroll
  for (int r = 0; r < 8; ++r) { mst[r] = -1e30f; lst[r] = 0.0f; }
  v8f o[4] = {v8f{}, v8f{}, v8f{}, v8f{}};   // d-chunks 0..3

  const float scale = 0.125f;   // 1/sqrt(64)
  const int NCH = SEQ / 32;

#if HAVE_TDM
  // Prologue: DMA chunk 0 into buffer 0 (wave 0 drives the TDM).
  if (wave == 0) {
    tdm_load_2d(lds_off_of(&Kbuf[0][0]), K, HD, SEQ, HD, 32, HD);
    tdm_load_2d(lds_off_of(&Vbuf[0][0]), V, SEQ, HD, 32, HD, SEQ);
  }
#endif

  for (int i = 0; i < NCH; ++i) {
    const int t0 = i * 32;
#if HAVE_TDM
    const int cur = i & 1, nxt = cur ^ 1;
    if (wave == 0) {
      if (i + 1 < NCH) {   // DMA next chunk while this one is consumed
        tdm_load_2d(lds_off_of(&Kbuf[nxt][0]), K + (size_t)(t0 + 32) * HD,
                    HD, SEQ, HD, 32, HD);
        tdm_load_2d(lds_off_of(&Vbuf[nxt][0]), V + (t0 + 32),
                    SEQ, HD, 32, HD, SEQ);
        __builtin_amdgcn_s_wait_tensorcnt(2);   // chunk i landed (in-order)
      } else {
        __builtin_amdgcn_s_wait_tensorcnt(0);
      }
    }
    __syncthreads();   // chunk i visible to both waves
#else
    const int cur = 0;
    {  // cooperative staging fallback: 4KB K + 4KB V via b128 copies
      const uint4* gk = (const uint4*)(K + (size_t)t0 * HD);
      uint4* sk = (uint4*)&Kbuf[0][0];
      for (int idx = threadIdx.x; idx < 256; idx += 64) sk[idx] = gk[idx];
      const __bf16* gv = V + t0;
      uint4* sv = (uint4*)&Vbuf[0][0];
      for (int idx = threadIdx.x; idx < 256; idx += 64) {
        int d = idx >> 2, c4 = idx & 3;             // 4 x uint4 per 32-el row
        sv[idx] = *(const uint4*)(gv + (size_t)d * SEQ + c4 * 8);
      }
      __syncthreads();
    }
#endif

    // ---- scores: two 16x16 tiles from LDS-resident K ----------------------
    const __bf16* Kc = &Kbuf[cur][0];
    v8f s0 = {}, s1 = {};
#pragma unroll
    for (int c = 0; c < 2; ++c) {
      v16bf b0 = *(const v16bf*)(Kc + l16 * HD + c * 32 + hl * 16);
      s0 = wmma_bf16(aq[c], b0, s0);
      v16bf b1 = *(const v16bf*)(Kc + (16 + l16) * HD + c * 32 + hl * 16);
      s1 = wmma_bf16(aq[c], b1, s1);
    }
    // ---- scale + additive mask -------------------------------------------
#pragma unroll
    for (int r = 0; r < 8; ++r) {
      int srow = m0 + r + hl * 8;
      s0[r] = s0[r] * scale + M[(size_t)srow * SEQ + t0 + l16];
      s1[r] = s1[r] * scale + M[(size_t)srow * SEQ + t0 + 16 + l16];
    }
    // ---- online softmax ---------------------------------------------------
#pragma unroll
    for (int r = 0; r < 8; ++r) {
      float cm = fmaxf(s0[r], s1[r]);
#pragma unroll
      for (int off = 1; off < 16; off <<= 1)
        cm = fmaxf(cm, __shfl_xor(cm, off));
      float mnew = fmaxf(mst[r], cm);
      float corr = __expf(mst[r] - mnew);
      float p0 = __expf(s0[r] - mnew);
      float p1 = __expf(s1[r] - mnew);
      float rs = p0 + p1;
#pragma unroll
      for (int off = 1; off < 16; off <<= 1)
        rs += __shfl_xor(rs, off);
      lst[r] = lst[r] * corr + rs;
      mst[r] = mnew;
#pragma unroll
      for (int c = 0; c < 4; ++c) o[c][r] *= corr;
      pls[wave][(r + hl * 8) * 32 + l16]      = (__bf16)p0;
      pls[wave][(r + hl * 8) * 32 + 16 + l16] = (__bf16)p1;
    }
    __syncthreads();
    // ---- reload P as a 16x32 A-fragment ----------------------------------
    v8bf plo = *(const v8bf*)(&pls[wave][l16 * 32 + hl * 16]);
    v8bf phi = *(const v8bf*)(&pls[wave][l16 * 32 + hl * 16 + 8]);
    v16bf ap = join8(plo, phi);
    // ---- O += P x V from LDS-resident V ----------------------------------
    const __bf16* Vc = &Vbuf[cur][0];
#pragma unroll
    for (int c = 0; c < 4; ++c) {
      int d = c * 16 + l16;
      v16bf bvv = *(const v16bf*)(Vc + d * 32 + hl * 16);
      o[c] = wmma_bf16(ap, bvv, o[c]);
    }
    __syncthreads();   // all reads of buffer `cur` done before it is re-filled
  }

  // ---- normalize and write out[s][b][h*64+d] ------------------------------
  float inv[8];
#pragma unroll
  for (int r = 0; r < 8; ++r) inv[r] = 1.0f / lst[r];
#pragma unroll
  for (int c = 0; c < 4; ++c) {
#pragma unroll
    for (int r = 0; r < 8; ++r) {
      int srow = m0 + r + hl * 8;
      int d    = c * 16 + l16;
      out[(size_t)srow * (BATCH * HID) + bidx * HID + hidx * HD + d] =
          o[c][r] * inv[r];
    }
  }
}

// ---------------------------------------------------------------------------
extern "C" void kernel_launch(void* const* d_in, const int* in_sizes, int n_in,
                              void* d_out, int out_size, void* d_ws,
                              size_t ws_size, hipStream_t stream) {
  const float* hs   = (const float*)d_in[0];
  const float* mask = (const float*)d_in[1];
  const float* Wq   = (const float*)d_in[2];
  const float* bq   = (const float*)d_in[3];
  const float* Wk   = (const float*)d_in[4];
  const float* bk   = (const float*)d_in[5];
  const float* Wv   = (const float*)d_in[6];
  const float* bv   = (const float*)d_in[7];
  float* out = (float*)d_out;

  char* w = (char*)d_ws;
  __bf16* Xbf = (__bf16*)w;  w += (size_t)MROWS * HID * 2;        // 8 MB
  __bf16* Wt3 = (__bf16*)w;  w += (size_t)3 * HID * HID * 2;      // 6 MB
  __bf16* Qb  = (__bf16*)w;  w += (size_t)NHEADS * SEQ * HD * 2;  // 8 MB
  __bf16* Kb  = (__bf16*)w;  w += (size_t)NHEADS * SEQ * HD * 2;  // 8 MB
  __bf16* Vt  = (__bf16*)w;  w += (size_t)NHEADS * HD * SEQ * 2;  // 8 MB

  const int nX = MROWS * HID;
  cvt_bf16_kernel<<<(nX + 255) / 256, 256, 0, stream>>>(hs, Xbf, nX);

  dim3 tb(32, 8), tg(HID / 32, HID / 32);
  transpose_cvt_kernel<<<tg, tb, 0, stream>>>(Wq, Wt3 + (size_t)0 * HID * HID);
  transpose_cvt_kernel<<<tg, tb, 0, stream>>>(Wk, Wt3 + (size_t)1 * HID * HID);
  transpose_cvt_kernel<<<tg, tb, 0, stream>>>(Wv, Wt3 + (size_t)2 * HID * HID);

  qkv_gemm_kernel<<<dim3(MROWS / 16, HID / 256, 3), 128, 0, stream>>>(
      Xbf, Wt3, bq, bk, bv, Qb, Kb, Vt);

  attn_kernel<<<dim3(SEQ / 32, NHEADS), 64, 0, stream>>>(Qb, Kb, Vt, mask, out);
}
